// RecurrentAE_14061722927625
// MI455X (gfx1250) — matrix operand order repaired and checked by reading, not compiled, not run on hardware
//
#include <hip/hip_runtime.h>
#include <math.h>

typedef __attribute__((ext_vector_type(16))) _Float16 v16h;
typedef __attribute__((ext_vector_type(8)))  _Float16 v8h;
typedef __attribute__((ext_vector_type(8)))  float    v8f;

#define Bsz 256
#define Tsz 512
#define Hsz 256
#define KT  8     // 256 / 32 k-slices per GEMM

// ---------------------------------------------------------------------------
// Prep kernel: convert w_hh [768,256] fp32 row-major into fp16 B-fragments in
// the per-lane WMMA layout:
//   frag index = ((nt*KT + kt)*32 + lane)*16 + i
//   lane n<16  holds column nt*16+n,     K = kt*32 + i       (i = 0..15)
//   lane n>=16 holds column nt*16+n-16,  K = kt*32 + 16 + i
// Each lane's 16 halfs (32B) are contiguous -> wide global load at use site.
// ---------------------------------------------------------------------------
__global__ void swizzle_whh_kernel(const float* __restrict__ whh,
                                   _Float16* __restrict__ bfrag) {
  int idx  = blockIdx.x * 256 + threadIdx.x;   // 0 .. 768*256-1
  int i    =  idx        & 15;
  int lane = (idx >> 4)  & 31;
  int ktnt =  idx >> 9;
  int kt   =  ktnt & 7;
  int nt   =  ktnt >> 3;                       // 0..47
  int col  = nt * 16 + (lane & 15);            // row j of w_hh (output col)
  int k    = kt * 32 + (lane & 16) + i;        // K index into h
  bfrag[idx] = (_Float16)whh[col * Hsz + k];
}

__device__ __forceinline__ float sigmf(float x) {
  return 1.0f / (1.0f + __expf(-x));
}
__device__ __forceinline__ float tanh_fast(float x) {
  float e = __expf(-2.0f * x);
  return (1.0f - e) / (1.0f + e);
}

// Preload this wave's 6 n-tiles x 8 k-slices of B into registers (384 VGPRs).
__device__ __forceinline__ void preload_B(const _Float16* __restrict__ bfrag,
                                          int wave, int lane, v16h B[3][2][8]) {
  #pragma unroll
  for (int g = 0; g < 3; ++g)
    #pragma unroll
    for (int s = 0; s < 2; ++s) {
      int nt = g * 16 + 2 * wave + s;
      #pragma unroll
      for (int kt = 0; kt < KT; ++kt)
        B[g][s][kt] = *(const v16h*)(bfrag + (size_t)((nt * KT + kt) * 32 + lane) * 16);
    }
}

__device__ __forceinline__ void load_consts(const float* __restrict__ wih_p,
                                            const float* __restrict__ bih_p,
                                            const float* __restrict__ bhh_p,
                                            int wave, int coll,
                                            float wih[3][2], float bih[3][2],
                                            float bhh[3][2]) {
  #pragma unroll
  for (int g = 0; g < 3; ++g)
    #pragma unroll
    for (int s = 0; s < 2; ++s) {
      int j3 = g * Hsz + (2 * wave + s) * 16 + coll;
      wih[g][s] = wih_p[j3];
      bih[g][s] = bih_p[j3];
      bhh[g][s] = bhh_p[j3];
    }
}

// dense head: xi[m] = sum_k h16[m][k]*wdense[k] + bdense; optional flipped store
__device__ __forceinline__ void dense_step(const _Float16* h16p,
                                           const float* __restrict__ wdense,
                                           float bd, float* xacc, float* xi_s,
                                           int tid, float* outp /* or null */) {
  if (tid < 16) xacc[tid] = 0.0f;
  __syncthreads();
  int m  = tid >> 4;
  int k0 = (tid & 15) * 16;
  float p = 0.0f;
  v8h hlo = *(const v8h*)(h16p + m * Hsz + k0);
  v8h hhi = *(const v8h*)(h16p + m * Hsz + k0 + 8);
  #pragma unroll
  for (int k = 0; k < 8; ++k) {
    p += (float)hlo[k] * wdense[k0 + k];
    p += (float)hhi[k] * wdense[k0 + 8 + k];
  }
  atomicAdd(&xacc[m], p);
  __syncthreads();
  if (tid < 16) {
    float v = xacc[tid] + bd;
    xi_s[tid] = v;
    if (outp) outp[tid * Tsz] = v;
  }
  __syncthreads();
}

// ---------------------------------------------------------------------------
// Main persistent kernel: 16 workgroups x 256 threads (8 waves).
// WG b handles batch rows [16b, 16b+16). Wave w owns gate columns
// [32w, 32w+32) of h for all 3 gates. Weights register-resident; h fp16 in
// LDS; encoder inputs staged to LDS once.
// ---------------------------------------------------------------------------
__global__ __launch_bounds__(256, 1)
void gru_ae_kernel(const float* __restrict__ x,
                   const _Float16* __restrict__ bfrag_enc,
                   const _Float16* __restrict__ bfrag_dec,
                   const float* __restrict__ wih_enc, const float* __restrict__ bih_enc,
                   const float* __restrict__ bhh_enc,
                   const float* __restrict__ wih_dec, const float* __restrict__ bih_dec,
                   const float* __restrict__ bhh_dec,
                   const float* __restrict__ wdense, const float* __restrict__ bdense,
                   float* __restrict__ out) {
  __shared__ __align__(16) _Float16 h16[16 * Hsz];   // 8 KB
  __shared__ __align__(16) float    xlds[16 * Tsz];  // 32 KB encoder inputs
  __shared__ float xi_s[16];
  __shared__ float xacc[16];

  const int tid    = threadIdx.x;
  const int wave   = tid >> 5;
  const int lane   = tid & 31;
  const int bm     = blockIdx.x * 16;        // first batch row of this WG
  const int rowoff = (lane & 16) >> 1;       // 0 or 8 (C/D row offset)
  const int coll   = lane & 15;              // C/D column within n-tile

  // Stage this WG's x block (contiguous [16 x 512] floats) into LDS.
  const float* xblk = x + (size_t)bm * Tsz;
  for (int i = tid; i < 16 * Tsz; i += 256) xlds[i] = xblk[i];

  // h = 0
  #pragma unroll
  for (int i = 0; i < 16; ++i) h16[tid * 16 + i] = (_Float16)0.0f;

  const float bd = bdense[0];

  // ======================= encoder =======================
  {
    float wih[3][2], bih[3][2], bhh[3][2];
    load_consts(wih_enc, bih_enc, bhh_enc, wave, coll, wih, bih, bhh);
    v16h B[3][2][KT];
    preload_B(bfrag_enc, wave, lane, B);
    __syncthreads();

    #pragma unroll 1
    for (int t = 0; t < Tsz; ++t) {
      v8f acc[3][2];
      #pragma unroll
      for (int g = 0; g < 3; ++g)
        #pragma unroll
        for (int s = 0; s < 2; ++s)
          #pragma unroll
          for (int v = 0; v < 8; ++v) acc[g][s][v] = 0.0f;

      #pragma unroll
      for (int kt = 0; kt < KT; ++kt) {
        const _Float16* ap = h16 + (lane & 15) * Hsz + kt * 32 + rowoff * 2;
        v8h alo = *(const v8h*)(ap);
        v8h ahi = *(const v8h*)(ap + 8);
        v16h a;
        #pragma unroll
        for (int i = 0; i < 8; ++i) { a[i] = alo[i]; a[i + 8] = ahi[i]; }
        #pragma unroll
        for (int g = 0; g < 3; ++g)
          #pragma unroll
          for (int s = 0; s < 2; ++s)
            acc[g][s] = __builtin_amdgcn_wmma_f32_16x16x32_f16(
                false, a, false, B[g][s][kt], (short)0, acc[g][s], false, false);
      }
      __syncthreads();   // all A-fragment reads of h16 complete

      #pragma unroll
      for (int s = 0; s < 2; ++s) {
        const int cb = (2 * wave + s) * 16 + coll;
        #pragma unroll
        for (int v = 0; v < 8; ++v) {
          int m = rowoff + v;
          float xt   = xlds[m * Tsz + t];
          float hold = (float)h16[m * Hsz + cb];
          float r = sigmf(xt * wih[0][s] + bih[0][s] + acc[0][s][v] + bhh[0][s]);
          float z = sigmf(xt * wih[1][s] + bih[1][s] + acc[1][s][v] + bhh[1][s]);
          float n = tanh_fast(xt * wih[2][s] + bih[2][s] +
                              r * (acc[2][s][v] + bhh[2][s]));
          h16[m * Hsz + cb] = (_Float16)((1.0f - z) * n + z * hold);
        }
      }
      __syncthreads();   // updated h visible to all waves
    }
  }

  // x0 = h_n @ w_dense^T + b_dense (no output emitted)
  dense_step(h16, wdense, bd, xacc, xi_s, tid, nullptr);

  // ======================= decoder =======================
  {
    float wih[3][2], bih[3][2], bhh[3][2];
    load_consts(wih_dec, bih_dec, bhh_dec, wave, coll, wih, bih, bhh);
    v16h B[3][2][KT];
    preload_B(bfrag_dec, wave, lane, B);
    __syncthreads();

    #pragma unroll 1
    for (int t = 0; t < Tsz; ++t) {
      v8f acc[3][2];
      #pragma unroll
      for (int g = 0; g < 3; ++g)
        #pragma unroll
        for (int s = 0; s < 2; ++s)
          #pragma unroll
          for (int v = 0; v < 8; ++v) acc[g][s][v] = 0.0f;

      #pragma unroll
      for (int kt = 0; kt < KT; ++kt) {
        const _Float16* ap = h16 + (lane & 15) * Hsz + kt * 32 + rowoff * 2;
        v8h alo = *(const v8h*)(ap);
        v8h ahi = *(const v8h*)(ap + 8);
        v16h a;
        #pragma unroll
        for (int i = 0; i < 8; ++i) { a[i] = alo[i]; a[i + 8] = ahi[i]; }
        #pragma unroll
        for (int g = 0; g < 3; ++g)
          #pragma unroll
          for (int s = 0; s < 2; ++s)
            acc[g][s] = __builtin_amdgcn_wmma_f32_16x16x32_f16(
                false, a, false, B[g][s][kt], (short)0, acc[g][s], false, false);
      }
      __syncthreads();

      #pragma unroll
      for (int s = 0; s < 2; ++s) {
        const int cb = (2 * wave + s) * 16 + coll;
        #pragma unroll
        for (int v = 0; v < 8; ++v) {
          int m = rowoff + v;
          float xt   = xi_s[m];
          float hold = (float)h16[m * Hsz + cb];
          float r = sigmf(xt * wih[0][s] + bih[0][s] + acc[0][s][v] + bhh[0][s]);
          float z = sigmf(xt * wih[1][s] + bih[1][s] + acc[1][s][v] + bhh[1][s]);
          float n = tanh_fast(xt * wih[2][s] + bih[2][s] +
                              r * (acc[2][s][v] + bhh[2][s]));
          h16[m * Hsz + cb] = (_Float16)((1.0f - z) * n + z * hold);
        }
      }
      __syncthreads();

      // x_{t+1} = h @ w_dense^T + b_dense ; output is time-flipped
      dense_step(h16, wdense, bd, xacc, xi_s, tid,
                 out + (size_t)bm * Tsz + (Tsz - 1 - t));
    }
  }
}

// ---------------------------------------------------------------------------
extern "C" void kernel_launch(void* const* d_in, const int* in_sizes, int n_in,
                              void* d_out, int out_size, void* d_ws, size_t ws_size,
                              hipStream_t stream) {
  const float* x        = (const float*)d_in[0];
  const float* w_ih_enc = (const float*)d_in[1];
  const float* w_hh_enc = (const float*)d_in[2];
  const float* b_ih_enc = (const float*)d_in[3];
  const float* b_hh_enc = (const float*)d_in[4];
  const float* w_ih_dec = (const float*)d_in[5];
  const float* w_hh_dec = (const float*)d_in[6];
  const float* b_ih_dec = (const float*)d_in[7];
  const float* b_hh_dec = (const float*)d_in[8];
  const float* w_dense  = (const float*)d_in[9];
  const float* b_dense  = (const float*)d_in[10];

  _Float16* bfrag_enc = (_Float16*)d_ws;
  _Float16* bfrag_dec = bfrag_enc + (size_t)3 * Hsz * Hsz;  // 768*256 halfs

  // one thread per swizzled element: 768*256 = 196608 = 768 blocks * 256
  swizzle_whh_kernel<<<768, 256, 0, stream>>>(w_hh_enc, bfrag_enc);
  swizzle_whh_kernel<<<768, 256, 0, stream>>>(w_hh_dec, bfrag_dec);

  gru_ae_kernel<<<Bsz / 16, 256, 0, stream>>>(
      x, bfrag_enc, bfrag_dec,
      w_ih_enc, b_ih_enc, b_hh_enc,
      w_ih_dec, b_ih_dec, b_hh_dec,
      w_dense, b_dense, (float*)d_out);
}
